// GraphormerBlock_24189255811169
// MI455X (gfx1250) — matrix operand order repaired and checked by reading, not compiled, not run on hardware
//
#include <hip/hip_runtime.h>
#include <hip/hip_bf16.h>

#define Nn 3072
#define Cc 512
#define Hh 8
#define Dd 64
#define Rr 8
#define Tt 6
#define Ee 49152
#define Ss 512
#define NBUK 21
#define C4 2048

typedef unsigned short ushort_t;
typedef __attribute__((ext_vector_type(16))) __bf16 v16bf;
typedef __attribute__((ext_vector_type(8)))  float  v8f;

struct alignas(16) B16 { unsigned long long a, b; };
union FragU { v16bf v; B16 h[2]; };

__device__ __forceinline__ ushort_t f2bf(float f) {
    unsigned u = __float_as_uint(f);
    unsigned r = u + 0x7FFFu + ((u >> 16) & 1u);
    return (ushort_t)(r >> 16);
}

// A-fragment (16x32 bf16): lanes 0-15 hold K chunks [0,8)+[16,24); lanes 16-31 hold [8,16)+[24,32)
__device__ __forceinline__ v16bf load_frag_a(const ushort_t* p, bool hiLane) {
    FragU u;
    const ushort_t* p0 = p + (hiLane ? 8 : 0);
    u.h[0] = *(const B16*)(p0);
    u.h[1] = *(const B16*)(p0 + 16);
    return u.v;
}

// B-fragment (32x16 bf16, NT: row n contracts last dim): lanes 0-15 K=[0,8)+[16,24), lanes 16-31 K=[8,16)+[24,32)
__device__ __forceinline__ v16bf load_frag_b(const ushort_t* p, bool hiLane) {
    FragU u;
    const ushort_t* p0 = p + (hiLane ? 8 : 0);
    u.h[0] = *(const B16*)(p0);
    u.h[1] = *(const B16*)(p0 + 16);
    return u.v;
}

// ---------------------------------------------------------------------------
// Generic NT GEMM: C[M,Nc] = A[M,K](bf16) * B[Nc,K](bf16)^T (+bias)(relu?)(+resid)
// block = 128 threads = 4 waves; wave tile 64x64; block tile 128x128
// __launch_bounds__(128,1): full VGPR file -> no accumulator spills
// ---------------------------------------------------------------------------
__global__ void __launch_bounds__(128, 1)
gemm_nt_bf16(const ushort_t* __restrict__ A, int lda,
             const ushort_t* __restrict__ Bm, int ldb,
             const float* __restrict__ bias,
             float* __restrict__ Cm, int ldc,
             const float* __restrict__ resid, int ldr,
             int M, int Nc, int K, int relu) {
    const int lane = threadIdx.x & 31;
    const int w = threadIdx.x >> 5;
    const int rowBase = blockIdx.y * 128 + (w >> 1) * 64;
    const int colBase = blockIdx.x * 128 + (w & 1) * 64;
    if (colBase >= Nc || rowBase >= M) return;   // wave-uniform
    const bool hi = lane >= 16;
    const int l15 = lane & 15;

    v8f zero = {0.f, 0.f, 0.f, 0.f, 0.f, 0.f, 0.f, 0.f};
    v8f acc[4][4];
#pragma unroll
    for (int i = 0; i < 4; ++i)
#pragma unroll
        for (int j = 0; j < 4; ++j) acc[i][j] = zero;

    const ushort_t* pa[4];
    const ushort_t* pb[4];
#pragma unroll
    for (int mt = 0; mt < 4; ++mt) pa[mt] = A + (size_t)(rowBase + mt * 16 + l15) * lda;
#pragma unroll
    for (int nt = 0; nt < 4; ++nt) pb[nt] = Bm + (size_t)(colBase + nt * 16 + l15) * ldb;

    for (int k = 0; k < K; k += 32) {
        v16bf af[4], bf_[4];
#pragma unroll
        for (int mt = 0; mt < 4; ++mt) af[mt] = load_frag_a(pa[mt] + k, hi);
#pragma unroll
        for (int nt = 0; nt < 4; ++nt) bf_[nt] = load_frag_b(pb[nt] + k, hi);
        if (k + 32 < K) {
            __builtin_prefetch(pa[0] + k + 32, 0, 1);
            __builtin_prefetch(pb[0] + k + 32, 0, 1);
        }
#pragma unroll
        for (int mt = 0; mt < 4; ++mt)
#pragma unroll
            for (int nt = 0; nt < 4; ++nt)
                acc[mt][nt] = __builtin_amdgcn_wmma_f32_16x16x32_bf16(
                    false, af[mt], false, bf_[nt], (short)0, acc[mt][nt], false, false);
    }

#pragma unroll
    for (int mt = 0; mt < 4; ++mt)
#pragma unroll
        for (int nt = 0; nt < 4; ++nt) {
            int n = colBase + nt * 16 + l15;
            float bv = bias ? bias[n] : 0.f;
#pragma unroll
            for (int r = 0; r < 8; ++r) {
                int m = rowBase + mt * 16 + r + (hi ? 8 : 0);
                float v = acc[mt][nt][r] + bv;
                if (relu) v = fmaxf(v, 0.f);
                if (resid) v += resid[(size_t)m * ldr + n];
                Cm[(size_t)m * ldc + n] = v;
            }
        }
}

// ---------------------------------------------------------------------------
// Attention scores: S[h,m,n] = (Q_h K_h^T)/8 + typepair[tt[m],tt[n],h]
// ---------------------------------------------------------------------------
__global__ void __launch_bounds__(128, 1)
scores_kernel(const ushort_t* __restrict__ qb,
              const ushort_t* __restrict__ kmat,
              const int* __restrict__ tt,
              const float* __restrict__ typepair,
              float* __restrict__ scores) {
    const int h = blockIdx.z;
    const int lane = threadIdx.x & 31;
    const int w = threadIdx.x >> 5;
    const int rowBase = blockIdx.y * 128 + (w >> 1) * 64;
    const int colBase = blockIdx.x * 128 + (w & 1) * 64;
    const bool hi = lane >= 16;
    const int l15 = lane & 15;

    v8f zero = {0.f, 0.f, 0.f, 0.f, 0.f, 0.f, 0.f, 0.f};
    v8f acc[4][4];
#pragma unroll
    for (int i = 0; i < 4; ++i)
#pragma unroll
        for (int j = 0; j < 4; ++j) acc[i][j] = zero;

#pragma unroll
    for (int ks = 0; ks < 64; ks += 32) {
        v16bf af[4], bf_[4];
#pragma unroll
        for (int mt = 0; mt < 4; ++mt)
            af[mt] = load_frag_a(qb + (size_t)(rowBase + mt * 16 + l15) * Cc + h * 64 + ks, hi);
#pragma unroll
        for (int nt = 0; nt < 4; ++nt)
            bf_[nt] = load_frag_b(kmat + (size_t)(colBase + nt * 16 + l15) * Cc + h * 64 + ks, hi);
#pragma unroll
        for (int mt = 0; mt < 4; ++mt)
#pragma unroll
            for (int nt = 0; nt < 4; ++nt)
                acc[mt][nt] = __builtin_amdgcn_wmma_f32_16x16x32_bf16(
                    false, af[mt], false, bf_[nt], (short)0, acc[mt][nt], false, false);
    }

    float* outh = scores + (size_t)h * Nn * Nn;
#pragma unroll
    for (int mt = 0; mt < 4; ++mt)
#pragma unroll
        for (int nt = 0; nt < 4; ++nt) {
            int n = colBase + nt * 16 + l15;
            int ttn = tt[n];
#pragma unroll
            for (int r = 0; r < 8; ++r) {
                int m = rowBase + mt * 16 + r + (hi ? 8 : 0);
                float v = acc[mt][nt][r] * 0.125f +
                          typepair[((size_t)tt[m] * Tt + ttn) * Hh + h];
                outh[(size_t)m * Nn + n] = v;
            }
        }
}

// ---------------------------------------------------------------------------
// Elementwise / reduction kernels
// ---------------------------------------------------------------------------
__global__ void f32_to_bf16_k(const float* __restrict__ in, ushort_t* __restrict__ out, size_t n) {
    size_t i = (size_t)blockIdx.x * blockDim.x + threadIdx.x;
    if (i < n) out[i] = f2bf(in[i]);
}

__global__ void transpose_bf16_k(const float* __restrict__ in, ushort_t* __restrict__ out,
                                 int rows, int cols) {
    __shared__ ushort_t tile[32][33];
    int c0 = blockIdx.x * 32, r0 = blockIdx.y * 32;
#pragma unroll
    for (int i = 0; i < 4; ++i) {
        int r = r0 + threadIdx.y + i * 8, c = c0 + threadIdx.x;
        if (r < rows && c < cols) tile[threadIdx.y + i * 8][threadIdx.x] = f2bf(in[(size_t)r * cols + c]);
    }
    __syncthreads();
#pragma unroll
    for (int i = 0; i < 4; ++i) {
        int r = r0 + threadIdx.x, c = c0 + threadIdx.y + i * 8;
        if (r < rows && c < cols) out[(size_t)c * rows + r] = tile[threadIdx.x][threadIdx.y + i * 8];
    }
}

__global__ void edge_bias_k(const int* __restrict__ src, const int* __restrict__ dst,
                            const int* __restrict__ erel, const float* __restrict__ adj,
                            float* __restrict__ scores) {
    int idx = blockIdx.x * blockDim.x + threadIdx.x;
    if (idx >= Ee * Hh) return;
    int e = idx >> 3, h = idx & 7;
    atomicAdd(&scores[(size_t)h * Nn * Nn + (size_t)src[e] * Nn + dst[e]],
              adj[erel[e] * Hh + h]);
}

__global__ void temp_bias_k(const int* __restrict__ seed_idx, const float* __restrict__ tvec,
                            const float* __restrict__ table, float* __restrict__ scores) {
    int idx = blockIdx.x * blockDim.x + threadIdx.x;
    if (idx >= Ss * Nn) return;
    int s = idx / Nn, n = idx - s * Nn;
    int row = seed_idx[s];
    float dt = tvec[n] - tvec[row];
    float sgn = (dt > 0.f) ? 1.f : ((dt < 0.f) ? -1.f : 0.f);
    float sl = sgn * log1pf(fabsf(dt) + 1e-6f);
    float norm = (fminf(fmaxf(sl, -5.f), 5.f) + 5.f) / (10.f + 1e-9f);
    int b = (int)floorf(norm * (float)(NBUK - 1));
    b = b < 0 ? 0 : (b > NBUK - 1 ? NBUK - 1 : b);
#pragma unroll
    for (int h = 0; h < Hh; ++h)
        scores[(size_t)h * Nn * Nn + (size_t)row * Nn + n] += table[b * Hh + h];
}

// Row softmax on fp32 scores, writes bf16 probs IN PLACE (row front half)
__global__ void __launch_bounds__(256)
softmax_k(float* __restrict__ scores) {
    const int row = blockIdx.x;        // H*N rows
    const int tid = threadIdx.x;       // 256 threads, N = 12 chunks
    float* rowf = scores + (size_t)row * Nn;
    ushort_t* rowh = (ushort_t*)rowf;
    __shared__ float s[256];

    float m = -3.4e38f;
    for (int j = tid; j < Nn; j += 256) m = fmaxf(m, rowf[j]);
    s[tid] = m;
    for (int st = 128; st > 0; st >>= 1) { __syncthreads(); if (tid < st) s[tid] = fmaxf(s[tid], s[tid + st]); }
    __syncthreads(); m = s[0]; __syncthreads();

    float sum = 0.f;
    for (int j = tid; j < Nn; j += 256) sum += __expf(rowf[j] - m);
    s[tid] = sum;
    for (int st = 128; st > 0; st >>= 1) { __syncthreads(); if (tid < st) s[tid] += s[tid + st]; }
    __syncthreads(); float inv = 1.f / s[0]; __syncthreads();

    for (int base = 0; base < Nn; base += 256) {
        float x = rowf[base + tid];
        __syncthreads();                       // all reads of this chunk complete
        rowh[base + tid] = f2bf(__expf(x - m) * inv);
        __syncthreads();
    }
}

// LayerNorm over C=512, input = a (+ b if non-null)
__global__ void __launch_bounds__(256)
ln_k(float* __restrict__ out, const float* __restrict__ a,
     const float* __restrict__ b, const float* __restrict__ g,
     const float* __restrict__ beta) {
    const int row = blockIdx.x, tid = threadIdx.x;
    __shared__ float s1[256], s2[256];
    const float* ar = a + (size_t)row * Cc;
    const float* br = b ? b + (size_t)row * Cc : nullptr;
    float v0 = ar[tid] + (br ? br[tid] : 0.f);
    float v1 = ar[tid + 256] + (br ? br[tid + 256] : 0.f);
    s1[tid] = v0 + v1;
    s2[tid] = v0 * v0 + v1 * v1;
    for (int st = 128; st > 0; st >>= 1) { __syncthreads(); if (tid < st) { s1[tid] += s1[tid + st]; s2[tid] += s2[tid + st]; } }
    __syncthreads();
    float mu = s1[0] * (1.f / Cc);
    float var = s2[0] * (1.f / Cc) - mu * mu;
    float inv = rsqrtf(var + 1e-5f);
    out[(size_t)row * Cc + tid]       = (v0 - mu) * inv * g[tid]       + beta[tid];
    out[(size_t)row * Cc + tid + 256] = (v1 - mu) * inv * g[tid + 256] + beta[tid + 256];
}

__global__ void rel_scatter_k(const int* __restrict__ src, const int* __restrict__ dst,
                              const int* __restrict__ erel, const float* __restrict__ Y,
                              float* __restrict__ hbuf, const float* __restrict__ alpha_p) {
    size_t idx = (size_t)blockIdx.x * blockDim.x + threadIdx.x;
    if (idx >= (size_t)Ee * Cc) return;
    int e = (int)(idx >> 9), c = (int)(idx & 511);
    float alpha = *alpha_p;
    atomicAdd(&hbuf[(size_t)dst[e] * Cc + c],
              alpha * Y[((size_t)erel[e] * Nn + src[e]) * Cc + c]);
}

// ---------------------------------------------------------------------------
extern "C" void kernel_launch(void* const* d_in, const int* in_sizes, int n_in,
                              void* d_out, int out_size, void* d_ws, size_t ws_size,
                              hipStream_t stream) {
    const float* x        = (const float*)d_in[0];
    const int*   tok      = (const int*)d_in[1];
    const int*   src      = (const int*)d_in[2];
    const int*   dst      = (const int*)d_in[3];
    const int*   erel     = (const int*)d_in[4];
    const int*   seed     = (const int*)d_in[5];
    const float* tvec     = (const float*)d_in[6];
    const float* Wq = (const float*)d_in[7];  const float* bq = (const float*)d_in[8];
    const float* Wk = (const float*)d_in[9];  const float* bk = (const float*)d_in[10];
    const float* Wv = (const float*)d_in[11]; const float* bv = (const float*)d_in[12];
    const float* Wo = (const float*)d_in[13]; const float* bo = (const float*)d_in[14];
    const float* W1 = (const float*)d_in[15]; const float* b1 = (const float*)d_in[16];
    const float* W2 = (const float*)d_in[17]; const float* b2 = (const float*)d_in[18];
    const float* ln1g = (const float*)d_in[19]; const float* ln1b = (const float*)d_in[20];
    const float* ln2g = (const float*)d_in[21]; const float* ln2b = (const float*)d_in[22];
    const float* adj  = (const float*)d_in[23];
    const float* typepair = (const float*)d_in[24];
    const float* ttable   = (const float*)d_in[25];
    const float* rel_w    = (const float*)d_in[26];
    const float* rel_b    = (const float*)d_in[27];
    const float* alpha_p  = (const float*)d_in[28];
    float* out = (float*)d_out;

    size_t cur = 0;
    auto alloc = [&](size_t bytes) -> void* {
        void* p = (char*)d_ws + cur;
        cur += (bytes + 255) & ~(size_t)255;
        return p;
    };
    ushort_t* xb   = (ushort_t*)alloc((size_t)Nn * Cc * 2);
    ushort_t* WqT  = (ushort_t*)alloc((size_t)Cc * Cc * 2);
    ushort_t* WkT  = (ushort_t*)alloc((size_t)Cc * Cc * 2);
    ushort_t* WvT  = (ushort_t*)alloc((size_t)Cc * Cc * 2);
    ushort_t* WoT  = (ushort_t*)alloc((size_t)Cc * Cc * 2);
    ushort_t* W1T  = (ushort_t*)alloc((size_t)Cc * C4 * 2);
    ushort_t* W2T  = (ushort_t*)alloc((size_t)Cc * C4 * 2);
    ushort_t* rwT  = (ushort_t*)alloc((size_t)Rr * Cc * Cc * 2);
    float*    q    = (float*)alloc((size_t)Nn * Cc * 4);
    float*    kmat = (float*)alloc((size_t)Nn * Cc * 4);
    float*    v    = (float*)alloc((size_t)Nn * Cc * 4);
    ushort_t* qb   = (ushort_t*)alloc((size_t)Nn * Cc * 2);
    ushort_t* kb   = (ushort_t*)alloc((size_t)Nn * Cc * 2);
    ushort_t* vt   = (ushort_t*)alloc((size_t)Cc * Nn * 2);
    float*    ctx  = (float*)alloc((size_t)Nn * Cc * 4);
    ushort_t* ctxb = (ushort_t*)alloc((size_t)Nn * Cc * 2);
    float*    ao   = (float*)alloc((size_t)Nn * Cc * 4);
    float*    hbuf = (float*)alloc((size_t)Nn * Cc * 4);
    ushort_t* hb   = (ushort_t*)alloc((size_t)Nn * Cc * 2);
    float*    scores = (float*)alloc((size_t)Hh * Nn * Nn * 4);
    // region reuse after ctx GEMM (stream-ordered): Y, f1, f1b, tmp alias scores
    char* big = (char*)scores;
    float*    Y   = (float*)(big);
    float*    f1  = (float*)(big + (size_t)Rr * Nn * Cc * 4);
    ushort_t* f1b = (ushort_t*)(big + (size_t)Rr * Nn * Cc * 4 + (size_t)Nn * C4 * 4);
    float*    tmp = (float*)(big + (size_t)Rr * Nn * Cc * 4 + (size_t)Nn * C4 * 4 + (size_t)Nn * C4 * 2);

    const dim3 tb(32, 8);
    // convert + transpose weights to bf16
    f32_to_bf16_k<<<(Nn * Cc + 255) / 256, 256, 0, stream>>>(x, xb, (size_t)Nn * Cc);
    transpose_bf16_k<<<dim3(Cc / 32, Cc / 32), tb, 0, stream>>>(Wq, WqT, Cc, Cc);
    transpose_bf16_k<<<dim3(Cc / 32, Cc / 32), tb, 0, stream>>>(Wk, WkT, Cc, Cc);
    transpose_bf16_k<<<dim3(Cc / 32, Cc / 32), tb, 0, stream>>>(Wv, WvT, Cc, Cc);
    transpose_bf16_k<<<dim3(Cc / 32, Cc / 32), tb, 0, stream>>>(Wo, WoT, Cc, Cc);
    transpose_bf16_k<<<dim3(C4 / 32, Cc / 32), tb, 0, stream>>>(W1, W1T, Cc, C4);
    transpose_bf16_k<<<dim3(Cc / 32, C4 / 32), tb, 0, stream>>>(W2, W2T, C4, Cc);
    for (int r = 0; r < Rr; ++r)
        transpose_bf16_k<<<dim3(Cc / 32, Cc / 32), tb, 0, stream>>>(
            rel_w + (size_t)r * Cc * Cc, rwT + (size_t)r * Cc * Cc, Cc, Cc);

    // QKV projections
    const dim3 gNC(Cc / 128, Nn / 128);
    gemm_nt_bf16<<<gNC, 128, 0, stream>>>(xb, Cc, WqT, Cc, bq, q, Cc, nullptr, 0, Nn, Cc, Cc, 0);
    gemm_nt_bf16<<<gNC, 128, 0, stream>>>(xb, Cc, WkT, Cc, bk, kmat, Cc, nullptr, 0, Nn, Cc, Cc, 0);
    gemm_nt_bf16<<<gNC, 128, 0, stream>>>(xb, Cc, WvT, Cc, bv, v, Cc, nullptr, 0, Nn, Cc, Cc, 0);
    f32_to_bf16_k<<<(Nn * Cc + 255) / 256, 256, 0, stream>>>(q, qb, (size_t)Nn * Cc);
    f32_to_bf16_k<<<(Nn * Cc + 255) / 256, 256, 0, stream>>>(kmat, kb, (size_t)Nn * Cc);
    transpose_bf16_k<<<dim3(Cc / 32, Nn / 32), tb, 0, stream>>>(v, vt, Nn, Cc);  // vt[c][n]

    // scores + biases + softmax (probs stored bf16 in-place)
    scores_kernel<<<dim3(Nn / 128, Nn / 128, Hh), 128, 0, stream>>>(qb, kb, tok, typepair, scores);
    edge_bias_k<<<(Ee * Hh + 255) / 256, 256, 0, stream>>>(src, dst, erel, adj, scores);
    temp_bias_k<<<(Ss * Nn + 255) / 256, 256, 0, stream>>>(seed, tvec, ttable, scores);
    softmax_k<<<Hh * Nn, 256, 0, stream>>>(scores);

    // ctx = probs @ V per head (A = bf16 probs inside fp32 score rows, lda = 2N)
    for (int h = 0; h < Hh; ++h) {
        const ushort_t* Ah = (const ushort_t*)scores + (size_t)h * 2 * Nn * Nn;
        gemm_nt_bf16<<<dim3(1, Nn / 128), 128, 0, stream>>>(
            Ah, 2 * Nn, vt + (size_t)h * Dd * Nn, Nn, nullptr,
            ctx + (size_t)h * Dd, Cc, nullptr, 0, Nn, Dd, Nn, 0);
    }
    f32_to_bf16_k<<<(Nn * Cc + 255) / 256, 256, 0, stream>>>(ctx, ctxb, (size_t)Nn * Cc);
    gemm_nt_bf16<<<gNC, 128, 0, stream>>>(ctxb, Cc, WoT, Cc, bo, ao, Cc, nullptr, 0, Nn, Cc, Cc, 0);
    ln_k<<<Nn, 256, 0, stream>>>(hbuf, x, ao, ln1g, ln1b);

    // relation message passing: Y_r = X @ rel_w[r] + rel_b[r]; scatter to dst
    for (int r = 0; r < Rr; ++r)
        gemm_nt_bf16<<<gNC, 128, 0, stream>>>(xb, Cc, rwT + (size_t)r * Cc * Cc, Cc,
                                              rel_b + (size_t)r * Cc, Y + (size_t)r * Nn * Cc, Cc,
                                              nullptr, 0, Nn, Cc, Cc, 0);
    rel_scatter_k<<<(unsigned)(((size_t)Ee * Cc + 255) / 256), 256, 0, stream>>>(
        src, dst, erel, Y, hbuf, alpha_p);

    // FFN + LN2
    f32_to_bf16_k<<<(Nn * Cc + 255) / 256, 256, 0, stream>>>(hbuf, hb, (size_t)Nn * Cc);
    gemm_nt_bf16<<<dim3(C4 / 128, Nn / 128), 128, 0, stream>>>(
        hb, Cc, W1T, Cc, b1, f1, C4, nullptr, 0, Nn, C4, Cc, 1);
    f32_to_bf16_k<<<(Nn * C4 + 255) / 256, 256, 0, stream>>>(f1, f1b, (size_t)Nn * C4);
    gemm_nt_bf16<<<gNC, 128, 0, stream>>>(f1b, C4, W2T, C4, b2, tmp, Cc, hbuf, Cc, Nn, Cc, C4, 0);
    ln_k<<<Nn, 256, 0, stream>>>(out, tmp, nullptr, ln2g, ln2b);
}
